// ConvexHullPointerNetwork_57303453663418
// MI455X (gfx1250) — matrix operand order repaired and checked by reading, not compiled, not run on hardware
//
#include <hip/hip_runtime.h>

// Problem constants (from reference): B=64, S=512, D_IN=2, H=512
#define BATCH 64
#define SEQ   512
#define HID   512
#define G4H   2048   // 4*H gate rows

typedef __attribute__((ext_vector_type(16))) __bf16 v16bf;
typedef __attribute__((ext_vector_type(8)))  float  v8f;

// ---------- bf16 helpers (raw ushort storage; RNE convert) ----------
__device__ __forceinline__ unsigned short f2bf(float f) {
    union { float f; unsigned u; } a; a.f = f;
    unsigned r = (a.u + 0x7FFFu + ((a.u >> 16) & 1u)) >> 16;
    return (unsigned short)r;
}
__device__ __forceinline__ float bf2f(unsigned short s) {
    union { unsigned u; float f; } o; o.u = ((unsigned)s) << 16;
    return o.f;
}
__device__ __forceinline__ float sigmoid_f(float x) {
    return 1.0f / (1.0f + __expf(-x));
}
// CDNA5 transcendental tanh; v_nop after per TRANS hazard rule (ISA 7.4/7.3)
__device__ __forceinline__ float tanh_f(float x) {
    float r;
    asm volatile("v_tanh_f32 %0, %1\n\tv_nop" : "=v"(r) : "v"(x));
    return r;
}

// ---------- WMMA fragment loaders (contiguous k order, same for A and B) ----------
// A: 16x32 bf16, row-major src. lane<16: row=lane, K=0..15 ; lane>=16: row=lane-16, K=16..31
__device__ __forceinline__ v16bf load_a_frag(const unsigned short* A, int lda, int row0, int k0) {
    int lane = threadIdx.x & 31;
    int r  = row0 + (lane & 15);
    int kb = k0 + ((lane >> 4) << 4);
    const unsigned short* p = A + (size_t)r * lda + kb;
    union { v16bf v; uint4 q[2]; } u;
    u.q[0] = *(const uint4*)(p);
    u.q[1] = *(const uint4*)(p + 8);
    return u.v;
}
// B: 32x16 bf16 where B[k][n] = W[n][k] (W row-major [rows][ldw]).
// per lane: 16 contiguous bf16 of W[n0 + (lane&15)][k0 + (lane>>4)*16 ...]
__device__ __forceinline__ v16bf load_b_frag(const unsigned short* W, int ldw, int n0, int k0) {
    int lane = threadIdx.x & 31;
    int r  = n0 + (lane & 15);
    int kb = k0 + ((lane >> 4) << 4);
    return *(const v16bf*)(W + (size_t)r * ldw + kb);
}
__device__ __forceinline__ v8f wmma_bf16(v16bf a, v16bf b, v8f c) {
    return __builtin_amdgcn_wmma_f32_16x16x32_bf16(false, a, false, b, (short)0, c, false, false);
}

// ---------- utility kernels ----------
__global__ __launch_bounds__(256) void k_f32_to_bf16(const float* __restrict__ s,
                                                     unsigned short* __restrict__ d, int n) {
    int i = blockIdx.x * 256 + threadIdx.x;
    if (i < n) d[i] = f2bf(s[i]);
}
__global__ __launch_bounds__(256) void k_zero_u16(unsigned short* p, int n) {
    int i = blockIdx.x * 256 + threadIdx.x;
    if (i < n) p[i] = 0;
}
__global__ __launch_bounds__(256) void k_zero_f32(float* p, int n) {
    int i = blockIdx.x * 256 + threadIdx.x;
    if (i < n) p[i] = 0.0f;
}

// ---------- encoder LSTM step: gates = h@Whh^T + x@Wih^T + b ; update h,c ----------
// grid 32 blocks (16 h-columns each), 128 threads (4 waves, wave = 16-row M tile).
// K-outer loop: one A fragment feeds all 4 gate WMMAs (4x fewer A-side loads).
__global__ __launch_bounds__(128) void k_enc_step(
    const float* __restrict__ inputs, const float* __restrict__ Wih,
    const float* __restrict__ bih, const float* __restrict__ bhh,
    const unsigned short* __restrict__ Whh,
    const unsigned short* __restrict__ h_in, unsigned short* __restrict__ h_out,
    float* __restrict__ c, unsigned short* __restrict__ enc_bf, int t)
{
    int lane = threadIdx.x & 31;
    int wave = threadIdx.x >> 5;
    int n0 = blockIdx.x * 16;
    int m0 = wave * 16;
    v8f z = {0,0,0,0,0,0,0,0};
    v8f acc[4] = {z, z, z, z};
    #pragma unroll 4
    for (int kt = 0; kt < 16; ++kt) {
        v16bf a = load_a_frag(h_in, HID, m0, kt * 32);
        #pragma unroll
        for (int g = 0; g < 4; ++g)
            acc[g] = wmma_bf16(a, load_b_frag(Whh, HID, g * HID + n0, kt * 32), acc[g]);
    }
    int n = n0 + (lane & 15);
    #pragma unroll
    for (int r = 0; r < 8; ++r) {
        int m = m0 + r + ((lane >> 4) << 3);
        float x0 = inputs[((size_t)m * SEQ + t) * 2 + 0];
        float x1 = inputs[((size_t)m * SEQ + t) * 2 + 1];
        float gv[4];
        #pragma unroll
        for (int g = 0; g < 4; ++g) {
            int gn = g * HID + n;
            gv[g] = acc[g][r] + bih[gn] + bhh[gn] + x0 * Wih[gn * 2 + 0] + x1 * Wih[gn * 2 + 1];
        }
        float ig = sigmoid_f(gv[0]);
        float fg = sigmoid_f(gv[1]);
        float gg = tanh_f(gv[2]);
        float og = sigmoid_f(gv[3]);
        float cn = fg * c[m * HID + n] + ig * gg;
        c[m * HID + n] = cn;
        unsigned short hb = f2bf(og * tanh_f(cn));
        h_out[m * HID + n] = hb;
        enc_bf[((size_t)m * SEQ + t) * HID + n] = hb;
    }
}

// ---------- keys = enc_outs @ W_key^T : [32768,512] x [512,512] ----------
// grid (2048, 2): blockIdx.x = 16-row M tile, blockIdx.y = group of 16 N tiles.
// Wave owns 4 N tiles; K-outer so one A fragment feeds 4 WMMAs.
__global__ __launch_bounds__(128) void k_keys_gemm(
    const unsigned short* __restrict__ enc_bf, const unsigned short* __restrict__ Wk,
    float* __restrict__ keys)
{
    int lane = threadIdx.x & 31;
    int wave = threadIdx.x >> 5;
    int m0 = blockIdx.x * 16;
    int ntb = blockIdx.y * 16 + wave * 4;   // first of 4 N tiles for this wave
    v8f z = {0,0,0,0,0,0,0,0};
    v8f acc[4] = {z, z, z, z};
    #pragma unroll 4
    for (int kt = 0; kt < 16; ++kt) {
        v16bf a = load_a_frag(enc_bf, HID, m0, kt * 32);
        #pragma unroll
        for (int j = 0; j < 4; ++j)
            acc[j] = wmma_bf16(a, load_b_frag(Wk, HID, (ntb + j) * 16, kt * 32), acc[j]);
    }
    #pragma unroll
    for (int j = 0; j < 4; ++j) {
        int n = (ntb + j) * 16 + (lane & 15);
        #pragma unroll
        for (int r = 0; r < 8; ++r) {
            int m = m0 + r + ((lane >> 4) << 3);
            keys[(size_t)m * HID + n] = acc[j][r];
        }
    }
}

// ---------- decoder LSTM step: gates = ctx@Wih^T + h@Whh^T + b ----------
__global__ __launch_bounds__(128) void k_dec_gates(
    const unsigned short* __restrict__ ctx,
    const unsigned short* __restrict__ Wih, const unsigned short* __restrict__ Whh,
    const float* __restrict__ bih, const float* __restrict__ bhh,
    const unsigned short* __restrict__ h_in, unsigned short* __restrict__ h_out,
    float* __restrict__ c)
{
    int lane = threadIdx.x & 31;
    int wave = threadIdx.x >> 5;
    int n0 = blockIdx.x * 16;
    int m0 = wave * 16;
    v8f z = {0,0,0,0,0,0,0,0};
    v8f acc[4] = {z, z, z, z};
    #pragma unroll 2
    for (int kt = 0; kt < 16; ++kt) {
        v16bf ax = load_a_frag(ctx,  HID, m0, kt * 32);
        v16bf ah = load_a_frag(h_in, HID, m0, kt * 32);
        #pragma unroll
        for (int g = 0; g < 4; ++g)
            acc[g] = wmma_bf16(ax, load_b_frag(Wih, HID, g * HID + n0, kt * 32), acc[g]);
        #pragma unroll
        for (int g = 0; g < 4; ++g)
            acc[g] = wmma_bf16(ah, load_b_frag(Whh, HID, g * HID + n0, kt * 32), acc[g]);
    }
    int n = n0 + (lane & 15);
    #pragma unroll
    for (int r = 0; r < 8; ++r) {
        int m = m0 + r + ((lane >> 4) << 3);
        float gv[4];
        #pragma unroll
        for (int g = 0; g < 4; ++g) {
            int gn = g * HID + n;
            gv[g] = acc[g][r] + bih[gn] + bhh[gn];
        }
        float ig = sigmoid_f(gv[0]);
        float fg = sigmoid_f(gv[1]);
        float gg = tanh_f(gv[2]);
        float og = sigmoid_f(gv[3]);
        float cn = fg * c[m * HID + n] + ig * gg;
        c[m * HID + n] = cn;
        h_out[m * HID + n] = f2bf(og * tanh_f(cn));
    }
}

// ---------- query = h @ W_query^T : [64,512] x [512,512] ----------
__global__ __launch_bounds__(128) void k_query_gemm(
    const unsigned short* __restrict__ h, const unsigned short* __restrict__ Wq,
    float* __restrict__ query)
{
    int lane = threadIdx.x & 31;
    int wave = threadIdx.x >> 5;
    int n0 = blockIdx.x * 16;
    int m0 = wave * 16;
    v8f acc = {0,0,0,0,0,0,0,0};
    #pragma unroll 4
    for (int kt = 0; kt < 16; ++kt) {
        acc = wmma_bf16(load_a_frag(h, HID, m0, kt * 32),
                        load_b_frag(Wq, HID, n0, kt * 32), acc);
    }
    int n = n0 + (lane & 15);
    #pragma unroll
    for (int r = 0; r < 8; ++r) {
        int m = m0 + r + ((lane >> 4) << 3);
        query[m * HID + n] = acc[r];
    }
}

// ---------- attention: scores -> out, softmax, context -> ctx ----------
// grid 64 blocks (one batch row each), 256 threads (8 waves)
__global__ __launch_bounds__(256) void k_dec_attn(
    const float* __restrict__ query, const float* __restrict__ keys,
    const float* __restrict__ vvec, const unsigned short* __restrict__ enc_bf,
    unsigned short* __restrict__ ctx, float* __restrict__ out, int t)
{
    __shared__ __align__(16) float qs[HID];
    __shared__ __align__(16) float vs[HID];
    __shared__ __align__(16) float sc[SEQ];
    __shared__ float red[8];
    int b = blockIdx.x;
    int tid = threadIdx.x;
    int lane = tid & 31, wave = tid >> 5;
    qs[tid]       = query[b * HID + tid];
    qs[tid + 256] = query[b * HID + tid + 256];
    vs[tid]       = vvec[tid];
    vs[tid + 256] = vvec[tid + 256];
    __syncthreads();

    const float* krow = keys + (size_t)b * SEQ * HID;
    const float4* qv4 = (const float4*)(&qs[lane * 16]);
    const float4* vv4 = (const float4*)(&vs[lane * 16]);
    for (int s = wave; s < SEQ; s += 8) {
        const float4* kp = (const float4*)(krow + (size_t)s * HID + lane * 16);
        if (s + 8 < SEQ)
            __builtin_prefetch(krow + (size_t)(s + 8) * HID + lane * 16, 0, 1);
        float sum = 0.0f;
        #pragma unroll
        for (int j = 0; j < 4; ++j) {
            float4 kv = kp[j];
            float4 qv = qv4[j];
            float4 vv = vv4[j];
            sum += tanh_f(qv.x + kv.x) * vv.x;
            sum += tanh_f(qv.y + kv.y) * vv.y;
            sum += tanh_f(qv.z + kv.z) * vv.z;
            sum += tanh_f(qv.w + kv.w) * vv.w;
        }
        #pragma unroll
        for (int o = 16; o > 0; o >>= 1) sum += __shfl_down(sum, o, 32);
        if (lane == 0) sc[s] = sum;
    }
    __syncthreads();

    // coalesced raw-score store (reference output is pre-softmax scores)
    float* orow = out + ((size_t)b * SEQ + t) * SEQ;
    for (int i = tid; i < SEQ; i += 256) orow[i] = sc[i];

    // block softmax over sc[0..511]
    float m = -3.402823466e38f;
    for (int i = tid; i < SEQ; i += 256) m = fmaxf(m, sc[i]);
    #pragma unroll
    for (int o = 16; o > 0; o >>= 1) m = fmaxf(m, __shfl_down(m, o, 32));
    if (lane == 0) red[wave] = m;
    __syncthreads();
    if (tid == 0) {
        float mm = red[0];
        for (int i = 1; i < 8; ++i) mm = fmaxf(mm, red[i]);
        red[0] = mm;
    }
    __syncthreads();
    float M = red[0];
    __syncthreads();
    float lsum = 0.0f;
    for (int i = tid; i < SEQ; i += 256) { float e = __expf(sc[i] - M); sc[i] = e; lsum += e; }
    #pragma unroll
    for (int o = 16; o > 0; o >>= 1) lsum += __shfl_down(lsum, o, 32);
    if (lane == 0) red[wave] = lsum;
    __syncthreads();
    if (tid == 0) {
        float ss = red[0];
        for (int i = 1; i < 8; ++i) ss += red[i];
        red[0] = 1.0f / ss;
    }
    __syncthreads();
    float inv = red[0];

    // context = probs @ enc_outs ; thread owns 2 adjacent hidden cols -> one b32/s
    const unsigned short* ep = enc_bf + (size_t)b * SEQ * HID + tid * 2;
    float a0 = 0.0f, a1 = 0.0f;
    for (int s = 0; s < SEQ; ++s) {
        float p = sc[s] * inv;
        unsigned pair = *(const unsigned*)(ep + (size_t)s * HID);
        a0 += p * bf2f((unsigned short)(pair & 0xFFFFu));
        a1 += p * bf2f((unsigned short)(pair >> 16));
    }
    ctx[b * HID + tid * 2 + 0] = f2bf(a0);
    ctx[b * HID + tid * 2 + 1] = f2bf(a1);
}

// ---------------------------------------------------------------------------
extern "C" void kernel_launch(void* const* d_in, const int* in_sizes, int n_in,
                              void* d_out, int out_size, void* d_ws, size_t ws_size,
                              hipStream_t stream) {
    (void)in_sizes; (void)n_in; (void)out_size; (void)ws_size;
    const float* inputs = (const float*)d_in[0];
    const float* eWih   = (const float*)d_in[1];
    const float* eWhh   = (const float*)d_in[2];
    const float* ebih   = (const float*)d_in[3];
    const float* ebhh   = (const float*)d_in[4];
    const float* dWih   = (const float*)d_in[5];
    const float* dWhh   = (const float*)d_in[6];
    const float* dbih   = (const float*)d_in[7];
    const float* dbhh   = (const float*)d_in[8];
    const float* Wk     = (const float*)d_in[9];
    const float* Wq     = (const float*)d_in[10];
    const float* vvec   = (const float*)d_in[11];
    float* out = (float*)d_out;

    char* ws = (char*)d_ws;
    size_t off = 0;
    auto take = [&](size_t bytes) -> void* {
        void* p = ws + off;
        off += (bytes + 255) & ~(size_t)255;
        return p;
    };
    unsigned short* eWhh_bf = (unsigned short*)take((size_t)G4H * HID * 2);
    unsigned short* dWih_bf = (unsigned short*)take((size_t)G4H * HID * 2);
    unsigned short* dWhh_bf = (unsigned short*)take((size_t)G4H * HID * 2);
    unsigned short* Wk_bf   = (unsigned short*)take((size_t)HID * HID * 2);
    unsigned short* Wq_bf   = (unsigned short*)take((size_t)HID * HID * 2);
    unsigned short* hbuf0   = (unsigned short*)take((size_t)BATCH * HID * 2);
    unsigned short* hbuf1   = (unsigned short*)take((size_t)BATCH * HID * 2);
    unsigned short* ctx     = (unsigned short*)take((size_t)BATCH * HID * 2);
    float*          cstate  = (float*)take((size_t)BATCH * HID * 4);
    float*          query   = (float*)take((size_t)BATCH * HID * 4);
    unsigned short* enc_bf  = (unsigned short*)take((size_t)BATCH * SEQ * HID * 2);
    float*          keys    = (float*)take((size_t)BATCH * SEQ * HID * 4);

    // weight conversions (f32 -> bf16, RNE)
    int nw = G4H * HID;
    k_f32_to_bf16<<<(nw + 255) / 256, 256, 0, stream>>>(eWhh, eWhh_bf, nw);
    k_f32_to_bf16<<<(nw + 255) / 256, 256, 0, stream>>>(dWih, dWih_bf, nw);
    k_f32_to_bf16<<<(nw + 255) / 256, 256, 0, stream>>>(dWhh, dWhh_bf, nw);
    int ns = HID * HID;
    k_f32_to_bf16<<<(ns + 255) / 256, 256, 0, stream>>>(Wk, Wk_bf, ns);
    k_f32_to_bf16<<<(ns + 255) / 256, 256, 0, stream>>>(Wq, Wq_bf, ns);

    // state init (h0 = c0 = 0; decoder x0 = 0)
    int nh = BATCH * HID;
    k_zero_u16<<<(nh + 255) / 256, 256, 0, stream>>>(hbuf0, nh);
    k_zero_u16<<<(nh + 255) / 256, 256, 0, stream>>>(ctx, nh);
    k_zero_f32<<<(nh + 255) / 256, 256, 0, stream>>>(cstate, nh);

    // encoder recurrence (ping-pong h to avoid cross-block RAW on h)
    for (int t = 0; t < SEQ; ++t) {
        unsigned short* hin  = (t & 1) ? hbuf1 : hbuf0;
        unsigned short* hout = (t & 1) ? hbuf0 : hbuf1;
        k_enc_step<<<32, 128, 0, stream>>>(inputs, eWih, ebih, ebhh, eWhh_bf,
                                           hin, hout, cstate, enc_bf, t);
    }

    // keys (step-invariant)
    {
        dim3 grid(BATCH * SEQ / 16, 2);
        k_keys_gemm<<<grid, 128, 0, stream>>>(enc_bf, Wk_bf, keys);
    }

    // decoder recurrence
    for (int u = 0; u < SEQ; ++u) {
        unsigned short* hin  = (u & 1) ? hbuf1 : hbuf0;
        unsigned short* hout = (u & 1) ? hbuf0 : hbuf1;
        k_dec_gates<<<32, 128, 0, stream>>>(ctx, dWih_bf, dWhh_bf, dbih, dbhh,
                                            hin, hout, cstate);
        k_query_gemm<<<32, 128, 0, stream>>>(hout, Wq_bf, query);
        k_dec_attn<<<BATCH, 256, 0, stream>>>(query, keys, vvec, enc_bf, ctx, out, u);
    }
}
